// CISSVAE_21569325760578
// MI455X (gfx1250) — matrix operand order repaired and checked by reading, not compiled, not run on hardware
//
#include <hip/hip_runtime.h>
#include <hip/hip_bf16.h>
#include <math.h>

// ---------------------------------------------------------------------------
// Types for WMMA fragments (CDNA5 / gfx1250, wave32)
// ---------------------------------------------------------------------------
typedef __attribute__((ext_vector_type(16))) __bf16       v16bf;
typedef __attribute__((ext_vector_type(8)))  float        v8f;
typedef __attribute__((ext_vector_type(4)))  unsigned int u32x4;

union BFrag { v16bf v; u32x4 q[2]; };
union U4x8  { u32x4 q; unsigned short s[8]; };

__device__ __forceinline__ unsigned short f32_to_bf16(float f) {
  unsigned int u = __float_as_uint(f);
  u += 0x7fffu + ((u >> 16) & 1u);   // round-to-nearest-even
  return (unsigned short)(u >> 16);
}

// Problem constants
constexpr int B_ROWS = 8192;
constexpr int D_IN   = 2048;
constexpr int H0     = 1024;
constexpr int H1     = 512;
constexpr int LAT    = 128;
constexpr int NCLUST = 8;

// Tiling: 128x128 block tile, 8 waves of 64x32, BK=32, double-buffered LDS.
constexpr int BM = 128, BN = 128, BK = 32;
constexpr int LDSS = 40;                      // shorts/LDS row: 32 data + 8 pad = 80B
constexpr int CAP  = B_ROWS + NCLUST * BM;    // 9216 padded rows (72 M-tiles)

// ---------------------------------------------------------------------------
// CDNA5 async global->LDS copy (ASYNCcnt) with guarded fallback.
// Builtin signature (from hipcc diagnostic): param1 = global ptr to
// GCC-vector int4 (addrspace(1), printed as __device__), param2 = LDS ptr,
// then imm offset, imm cpol.
// ---------------------------------------------------------------------------
#if __has_builtin(__builtin_amdgcn_global_load_async_to_lds_b128) && \
    __has_builtin(__builtin_amdgcn_s_wait_asynccnt)
#define HAVE_ASYNC_LDS 1
typedef int v4i_gcc __attribute__((vector_size(16)));
typedef __attribute__((address_space(1))) v4i_gcc as1_v4i;
typedef __attribute__((address_space(3))) v4i_gcc as3_v4i;
#endif

__device__ __forceinline__ void g2l_b128(const unsigned short* g, unsigned short* l) {
#ifdef HAVE_ASYNC_LDS
  __builtin_amdgcn_global_load_async_to_lds_b128((as1_v4i*)g, (as3_v4i*)l, 0, 0);
#else
  *(u32x4*)l = *(const u32x4*)g;
#endif
}

__device__ __forceinline__ void async_drain() {
#ifdef HAVE_ASYNC_LDS
  __builtin_amdgcn_s_wait_asynccnt(0);
#endif
}

// ---------------------------------------------------------------------------
// 1) Counting-sort rows by cluster; padded segment table + gather map.
// ---------------------------------------------------------------------------
__global__ void build_perm_kernel(const int* __restrict__ labels,
                                  int* __restrict__ meta,
                                  int* __restrict__ row_of_slot) {
  __shared__ int cnt[NCLUST], start[NCLUST + 1], cur[NCLUST];
  const int t = threadIdx.x;
  if (t < NCLUST) { cnt[t] = 0; cur[t] = 0; }
  __syncthreads();
  for (int i = t; i < B_ROWS; i += blockDim.x)
    atomicAdd(&cnt[labels[i] & (NCLUST - 1)], 1);
  __syncthreads();
  if (t == 0) {
    int o = 0;
    for (int c = 0; c < NCLUST; ++c) {
      start[c] = o;
      o += ((cnt[c] + BM - 1) / BM) * BM;   // pad each segment to M-tile
    }
    start[NCLUST] = o;
    for (int c = 0; c <= NCLUST; ++c) meta[c] = start[c];
  }
  __syncthreads();
  for (int i = t; i < CAP; i += blockDim.x) row_of_slot[i] = -1;
  __syncthreads();
  for (int i = t; i < B_ROWS; i += blockDim.x) {
    int c = labels[i] & (NCLUST - 1);
    int s = start[c] + atomicAdd(&cur[c], 1);
    row_of_slot[s] = i;
  }
}

// ---------------------------------------------------------------------------
// 2) Weight convert+transpose: f32 [C,K,N] -> bf16 [C,N,K] (one-time stream)
// ---------------------------------------------------------------------------
__global__ void wconv_kernel(const float* __restrict__ W,
                             unsigned short* __restrict__ Wt,
                             int K, int N, long total) {
  long i = (long)blockIdx.x * blockDim.x + threadIdx.x;
  if (i >= total) return;
  long kn = (long)K * N;
  long c  = i / kn;
  long r  = i - c * kn;
  int  k  = (int)(r / N);
  int  n  = (int)(r - (long)k * N);
  float v = __builtin_nontemporal_load(&W[i]);
  __builtin_nontemporal_store(f32_to_bf16(v), &Wt[c * kn + (long)n * K + k]);
}

// ---------------------------------------------------------------------------
// 3) Gather x rows into cluster-grouped order, f32 -> bf16. Pad slots -> 0.
// ---------------------------------------------------------------------------
__global__ void gather_x_kernel(const float* __restrict__ x,
                                const int* __restrict__ ros,
                                unsigned short* __restrict__ xg, int K) {
  const int s   = blockIdx.y;
  const int col = (blockIdx.x * blockDim.x + threadIdx.x) * 8;
  if (col >= K) return;
  const int r = ros[s];
  U4x8 o;
  if (r < 0) {
    for (int j = 0; j < 8; ++j) o.s[j] = 0;
  } else {
    const float* p = x + (size_t)r * K + col;
    for (int j = 0; j < 8; ++j) o.s[j] = f32_to_bf16(__builtin_nontemporal_load(&p[j]));
  }
  *(u32x4*)&xg[(size_t)s * K + col] = o.q;
}

// ---------------------------------------------------------------------------
// 4) Reparameterize: z[slot] = mu + eps * exp(0.5*logvar) (grouped bf16)
// ---------------------------------------------------------------------------
__global__ void zrep_kernel(const float* __restrict__ mu_rows,
                            const float* __restrict__ lv_rows,
                            const float* __restrict__ eps,
                            const int* __restrict__ ros,
                            unsigned short* __restrict__ z) {
  const int s = blockIdx.x;
  const int i = threadIdx.x;               // 128 threads
  const int r = ros[s];
  float v = 0.0f;
  if (r >= 0) {
    size_t idx = (size_t)r * LAT + i;
    v = mu_rows[idx] + eps[idx] * expf(0.5f * lv_rows[idx]);
  }
  z[(size_t)s * LAT + i] = f32_to_bf16(v);
}

// ---------------------------------------------------------------------------
// 5) bf16 WMMA GEMM:  out = [relu](A[M,K] @ Wt[N,K]^T + bias)
//    128x128x32 block tile, 8 waves (2 Mx4 N), 8 WMMAs/wave/K-step,
//    double-buffered LDS (ping-pong, one barrier/iter), async g->LDS copies.
// ---------------------------------------------------------------------------
template<bool UNSHARED, bool RELU, bool F32OUT>
__global__ __launch_bounds__(256)
void gemm_bf16_kernel(const unsigned short* __restrict__ A, int K,
                      const unsigned short* __restrict__ Wt,
                      const float* __restrict__ bias, int N,
                      unsigned short* __restrict__ outBf,
                      float* __restrict__ outF, int outStride,
                      const int* __restrict__ meta,
                      const int* __restrict__ ros) {
  __shared__ unsigned short lA[2][BM * LDSS];
  __shared__ unsigned short lB[2][BN * LDSS];
  __shared__ int rows_ld[BM];

  const int tid  = threadIdx.x;
  const int lane = tid & 31;
  const int wave = tid >> 5;
  const int wm   = wave >> 2;          // 0..1 : 64-row half of the block tile
  const int wn   = wave & 3;           // 0..3 : 32-col slab
  const int m0   = blockIdx.y * BM;
  const int n0   = blockIdx.x * BN;

  int cluster = 0;
  if (UNSHARED) {
    for (int c = 1; c < NCLUST; ++c)
      if (m0 >= meta[c]) cluster = c;
  }
  const unsigned short* Wp  = Wt + (size_t)cluster * N * K;
  const float*          bptr = bias + (size_t)(UNSHARED ? cluster : 0) * N;

  if (F32OUT) {
    if (tid < BM) rows_ld[tid] = ros[m0 + tid];
  }

  // Cooperative fill: 512 16B-chunks per matrix tile, 2 per thread.
  auto fill = [&](int it, int buf) {
    const int k0 = it * BK;
#pragma unroll
    for (int h = 0; h < 2; ++h) {
      const int e   = tid + h * 256;     // 0..511
      const int row = e >> 2;            // 0..127
      const int col = (e & 3) * 8;       // 0,8,16,24 shorts
      g2l_b128(&A [(size_t)(m0 + row) * K + k0 + col], &lA[buf][row * LDSS + col]);
      g2l_b128(&Wp[(size_t)(n0 + row) * K + k0 + col], &lB[buf][row * LDSS + col]);
    }
  };

  v8f acc[4][2] = {};

  const int kh = (lane >> 4) * 8;      // A-frag K-half (ISA 16-bit A layout)
  const int kb = (lane >> 4) * 8;      // B-frag K-half (same k order as A)
  const int l15 = lane & 15;

  const int nk = K / BK;
  fill(0, 0);
  async_drain();
  __syncthreads();

  for (int it = 0; it < nk; ++it) {
    const int cur = it & 1;
    if (it + 1 < nk) fill(it + 1, cur ^ 1);

    BFrag b0, b1;
    {
      const int r0 = (wn * 32 + l15) * LDSS;
      b0.q[0] = *(const u32x4*)&lB[cur][r0 + kb];
      b0.q[1] = *(const u32x4*)&lB[cur][r0 + kb + 16];
      const int r1 = (wn * 32 + 16 + l15) * LDSS;
      b1.q[0] = *(const u32x4*)&lB[cur][r1 + kb];
      b1.q[1] = *(const u32x4*)&lB[cur][r1 + kb + 16];
    }
#pragma unroll
    for (int mi = 0; mi < 4; ++mi) {
      const int r = (wm * 64 + mi * 16 + l15) * LDSS;
      BFrag a;
      a.q[0] = *(const u32x4*)&lA[cur][r + kh];
      a.q[1] = *(const u32x4*)&lA[cur][r + kh + 16];
      acc[mi][0] = __builtin_amdgcn_wmma_f32_16x16x32_bf16(false, a.v, false, b0.v,
                                                           (short)0, acc[mi][0], false, false);
      acc[mi][1] = __builtin_amdgcn_wmma_f32_16x16x32_bf16(false, a.v, false, b1.v,
                                                           (short)0, acc[mi][1], false, false);
    }

    async_drain();        // next tile fully landed in LDS (async path)
    __syncthreads();      // all waves done reading `cur`, next buffer ready
  }

  // Epilogue. C/D layout: lane l, VGPR j -> M = (l>>4)*8 + j, N = l&15.
  const int nc0   = n0 + wn * 32 + l15;
  const float bb0 = bptr[nc0];
  const float bb1 = bptr[nc0 + 16];

#pragma unroll
  for (int mi = 0; mi < 4; ++mi) {
    const int mbase = wm * 64 + mi * 16 + (lane >> 4) * 8;
#pragma unroll
    for (int j = 0; j < 8; ++j) {
      float v0 = acc[mi][0][j] + bb0;
      float v1 = acc[mi][1][j] + bb1;
      if (RELU) { v0 = fmaxf(v0, 0.f); v1 = fmaxf(v1, 0.f); }
      if (F32OUT) {
        const int r = rows_ld[mbase + j];
        if (r >= 0) {
          __builtin_nontemporal_store(v0, &outF[(size_t)r * outStride + nc0]);
          __builtin_nontemporal_store(v1, &outF[(size_t)r * outStride + nc0 + 16]);
        }
      } else {
        const size_t row = (size_t)(m0 + mbase + j);
        outBf[row * N + nc0]      = f32_to_bf16(v0);
        outBf[row * N + nc0 + 16] = f32_to_bf16(v1);
      }
    }
  }
}

// ---------------------------------------------------------------------------
// Launcher
// ---------------------------------------------------------------------------
extern "C" void kernel_launch(void* const* d_in, const int* in_sizes, int n_in,
                              void* d_out, int out_size, void* d_ws, size_t ws_size,
                              hipStream_t stream) {
  const float* x        = (const float*)d_in[0];
  const int*   labels   = (const int*)  d_in[1];
  const float* eps      = (const float*)d_in[2];
  const float* W_enc0   = (const float*)d_in[3];
  const float* b_enc0   = (const float*)d_in[4];
  const float* W_enc1   = (const float*)d_in[5];
  const float* b_enc1   = (const float*)d_in[6];
  const float* W_mu     = (const float*)d_in[7];
  const float* b_mu     = (const float*)d_in[8];
  const float* W_lv     = (const float*)d_in[9];
  const float* b_lv     = (const float*)d_in[10];
  const float* W_dec0   = (const float*)d_in[11];
  const float* b_dec0   = (const float*)d_in[12];
  const float* W_dec1   = (const float*)d_in[13];
  const float* b_dec1   = (const float*)d_in[14];
  const float* W_out    = (const float*)d_in[15];
  const float* b_out    = (const float*)d_in[16];

  float* out = (float*)d_out;
  const size_t RECON = (size_t)B_ROWS * D_IN;
  const size_t MUSZ  = (size_t)B_ROWS * LAT;
  float* out_mu = out + RECON;
  float* out_lv = out + RECON + MUSZ;

  // workspace carve-up (256B aligned regions)
  char*  ws  = (char*)d_ws;
  size_t off = 0;
  auto carve = [&](size_t bytes) -> char* {
    char* p = ws + off;
    off = (off + bytes + 255) & ~(size_t)255;
    return p;
  };
  int*            meta    = (int*)carve(64 * sizeof(int));
  int*            ros     = (int*)carve((size_t)CAP * sizeof(int));
  unsigned short* xg      = (unsigned short*)carve((size_t)CAP * D_IN * 2);
  unsigned short* h0      = (unsigned short*)carve((size_t)CAP * H0 * 2);
  unsigned short* h1      = (unsigned short*)carve((size_t)CAP * H1 * 2);
  unsigned short* zb      = (unsigned short*)carve((size_t)CAP * LAT * 2);
  unsigned short* d0      = (unsigned short*)carve((size_t)CAP * H1 * 2);
  unsigned short* d1      = (unsigned short*)carve((size_t)CAP * H0 * 2);
  unsigned short* Wt_enc0 = (unsigned short*)carve((size_t)NCLUST * D_IN * H0 * 2);
  unsigned short* Wt_enc1 = (unsigned short*)carve((size_t)H0 * H1 * 2);
  unsigned short* Wt_mu   = (unsigned short*)carve((size_t)H1 * LAT * 2);
  unsigned short* Wt_lv   = (unsigned short*)carve((size_t)H1 * LAT * 2);
  unsigned short* Wt_dec0 = (unsigned short*)carve((size_t)LAT * H1 * 2);
  unsigned short* Wt_dec1 = (unsigned short*)carve((size_t)NCLUST * H1 * H0 * 2);
  unsigned short* Wt_out  = (unsigned short*)carve((size_t)H0 * D_IN * 2);

  // 1) segment table + gather map
  build_perm_kernel<<<1, 256, 0, stream>>>(labels, meta, ros);

  // 2) weight convert+transpose (f32 [C,K,N] -> bf16 [C,N,K])
  auto wconv = [&](const float* src, unsigned short* dst, int C, int K, int N) {
    long total = (long)C * K * N;
    int  grid  = (int)((total + 255) / 256);
    wconv_kernel<<<grid, 256, 0, stream>>>(src, dst, K, N, total);
  };
  wconv(W_enc0, Wt_enc0, NCLUST, D_IN, H0);
  wconv(W_enc1, Wt_enc1, 1,      H0,   H1);
  wconv(W_mu,   Wt_mu,   1,      H1,   LAT);
  wconv(W_lv,   Wt_lv,   1,      H1,   LAT);
  wconv(W_dec0, Wt_dec0, 1,      LAT,  H1);
  wconv(W_dec1, Wt_dec1, NCLUST, H1,   H0);
  wconv(W_out,  Wt_out,  1,      H0,   D_IN);

  // 3) gather x into grouped bf16
  gather_x_kernel<<<dim3(D_IN / (256 * 8), CAP), 256, 0, stream>>>(x, ros, xg, D_IN);

  const int MB = CAP / BM;  // 72 M-tiles

  // 4) encoder layer 0 (unshared) : xg[CAP,2048] -> h0[CAP,1024], relu
  gemm_bf16_kernel<true, true, false><<<dim3(H0 / BN, MB), 256, 0, stream>>>(
      xg, D_IN, Wt_enc0, b_enc0, H0, h0, nullptr, 0, meta, ros);

  // 5) encoder layer 1 (shared) : h0 -> h1[CAP,512], relu
  gemm_bf16_kernel<false, true, false><<<dim3(H1 / BN, MB), 256, 0, stream>>>(
      h0, H0, Wt_enc1, b_enc1, H1, h1, nullptr, 0, meta, ros);

  // 6) latent heads (shared, no relu) : scatter f32 rows into d_out
  gemm_bf16_kernel<false, false, true><<<dim3(LAT / BN, MB), 256, 0, stream>>>(
      h1, H1, Wt_mu, b_mu, LAT, nullptr, out_mu, LAT, meta, ros);
  gemm_bf16_kernel<false, false, true><<<dim3(LAT / BN, MB), 256, 0, stream>>>(
      h1, H1, Wt_lv, b_lv, LAT, nullptr, out_lv, LAT, meta, ros);

  // 7) reparameterize -> grouped bf16 z
  zrep_kernel<<<CAP, LAT, 0, stream>>>(out_mu, out_lv, eps, ros, zb);

  // 8) decoder layer 0 (shared) : z[CAP,128] -> d0[CAP,512], relu
  gemm_bf16_kernel<false, true, false><<<dim3(H1 / BN, MB), 256, 0, stream>>>(
      zb, LAT, Wt_dec0, b_dec0, H1, d0, nullptr, 0, meta, ros);

  // 9) decoder layer 1 (unshared) : d0 -> d1[CAP,1024], relu
  gemm_bf16_kernel<true, true, false><<<dim3(H0 / BN, MB), 256, 0, stream>>>(
      d0, H1, Wt_dec1, b_dec1, H0, d1, nullptr, 0, meta, ros);

  // 10) output layer (shared, no relu) : scatter f32 recon rows into d_out
  gemm_bf16_kernel<false, false, true><<<dim3(D_IN / BN, MB), 256, 0, stream>>>(
      d1, H0, Wt_out, b_out, D_IN, nullptr, out, D_IN, meta, ros);
}
